// CondLaneRNNHead_26946624815740
// MI455X (gfx1250) — matrix operand order repaired and checked by reading, not compiled, not run on hardware
//
#include <hip/hip_runtime.h>
#include <hip/hip_bf16.h>

// ---------------------------------------------------------------------------
// CondInst-style dynamic mask head, fused 3-layer MLP over pixels.
//   x:                [4, 64, 80, 200] fp32
//   mask_head_params: [32, 8513] fp32   (w0 64x66 | w1 64x64 | w2 1x64 | b0 | b1 | b2)
//   out:              [32, 16000] fp32  (reshaped [1, 32, 80, 200])
// bf16 WMMA (f32 accumulate) for layers 0/1; coordinate channels + biases
// folded into fp32 accumulator init; layer2 as fp32 dot. 64 pixels per block
// (4 WMMA N-tiles) to amortize staging/barriers and reuse weight fragments.
// ---------------------------------------------------------------------------

typedef unsigned short u16_t;
typedef unsigned int   u32_t;
typedef __bf16 bf16_t;
typedef bf16_t v16bf __attribute__((ext_vector_type(16)));
typedef bf16_t v8bf  __attribute__((ext_vector_type(8)));
typedef float  v8f   __attribute__((ext_vector_type(8)));

#define HH 80
#define WW 200
#define HW 16000
#define CH 64
#define MI 32
#define NP 8513
#define NPIX 64            // pixels per block = 4 WMMA N-tiles

// ---------------------------------------------------------------------------
// Pre-pass: repack per-instance weights into a WMMA A-fragment lane layout
// (k contiguous per lane half, not the ISA table order) as bf16, plus fp32 bias / coord-weight /
// w2 arrays.  w0A/w1A: [m][kb(2)][mtile(4)][lane(32)][j(16)] bf16, j = 2v+p,
//   K(j,half) = 16*half + j + 32*kb, Mrow = 16*mt+lrow
// ---------------------------------------------------------------------------
__global__ void pack_weights_kernel(const float* __restrict__ p,
                                    bf16_t* __restrict__ w0A, bf16_t* __restrict__ w1A,
                                    float* __restrict__ b0a, float* __restrict__ wxa,
                                    float* __restrict__ wya, float* __restrict__ b1a,
                                    float* __restrict__ w2a, float* __restrict__ b2a) {
  const int m = blockIdx.x;
  const float* pm = p + (size_t)m * NP;
  const int t = threadIdx.x;

  for (int i = t; i < 4096; i += 256) {
    const int j    = i & 15;
    const int lane = (i >> 4) & 31;
    const int mt   = (i >> 9) & 3;
    const int kb   = i >> 11;
    const int half = lane >> 4;
    const int lrow = lane & 15;
    const int v    = j >> 1;
    const int pb   = j & 1;
    const int K    = 16 * half + 2 * v + pb + 32 * kb;
    const int Mr   = 16 * mt + lrow;
    // layer0: skip the 2 coordinate columns (handled in accumulator init)
    w0A[(size_t)m * 4096 + i] = (bf16_t)pm[Mr * 66 + K + 2];
    w1A[(size_t)m * 4096 + i] = (bf16_t)pm[4224 + Mr * 64 + K];
  }
  if (t < 64) {
    b0a[m * 64 + t] = pm[8384 + t];
    wxa[m * 64 + t] = pm[t * 66 + 0];   // weight on x-coordinate channel
    wya[m * 64 + t] = pm[t * 66 + 1];   // weight on y-coordinate channel
    b1a[m * 64 + t] = pm[8448 + t];
    w2a[m * 64 + t] = pm[8320 + t];
  }
  if (t == 0) b2a[m] = pm[8512] - 2.19f;  // mask bias shift
}

// ---------------------------------------------------------------------------
// Main fused kernel. Grid: (250 pixel blocks, 32 instances). Block: 128 (4
// waves); wave w owns output-channel tile w (rows 16w..16w+15); 4 N-tiles of
// 16 pixels each processed with shared A fragments / bias registers.
// ---------------------------------------------------------------------------
__global__ void __launch_bounds__(128)
cond_mask_head_kernel(const float* __restrict__ x, float* __restrict__ out,
                      const bf16_t* __restrict__ w0A, const bf16_t* __restrict__ w1A,
                      const float* __restrict__ b0a, const float* __restrict__ wxa,
                      const float* __restrict__ wya, const float* __restrict__ b1a,
                      const float* __restrict__ w2a, const float* __restrict__ b2a) {
  // transposed activation staging: [pixel(64)][channel, padded to 80] bf16
  __shared__ bf16_t xsT[NPIX * 80];
  __shared__ bf16_t hsT[NPIX * 80];
  __shared__ float  hs2[NPIX * 72];   // layer1 output fp32 for the final dot

  const int m    = blockIdx.y;
  const int n    = m >> 3;               // num_ins = 8 -> image index
  const int pix0 = blockIdx.x * NPIX;
  const int t    = threadIdx.x;

  // ---- stage x tile (64 ch x 64 px) into LDS, transposed, as bf16 ----------
  {
    const int c = t >> 1;
    const int g = t & 1;                 // which 32-pixel half
    const float* src = x + ((size_t)(n * CH + c)) * HW + pix0 + g * 32;
#pragma unroll
    for (int q = 0; q < 8; ++q) {
      const float4 f = *(const float4*)(src + 4 * q);
      const int pb = g * 32 + 4 * q;
      xsT[(pb + 0) * 80 + c] = (bf16_t)f.x;
      xsT[(pb + 1) * 80 + c] = (bf16_t)f.y;
      xsT[(pb + 2) * 80 + c] = (bf16_t)f.z;
      xsT[(pb + 3) * 80 + c] = (bf16_t)f.w;
    }
  }
  __syncthreads();

  const int wave = t >> 5;
  const int lane = t & 31;
  const int half = lane >> 4;
  const int lrow = lane & 15;           // N column within a tile
  const int mt   = wave;                // output-channel tile
  const int Mb   = 16 * mt + 8 * half;  // first of this lane's 8 C/D rows

  // per-lane bias / coordinate-weight registers (shared across all 4 N-tiles)
  float b0v[8], wxv[8], wyv[8], b1v[8];
  {
    const float* bp = b0a + m * 64 + Mb;
    const float* xp = wxa + m * 64 + Mb;
    const float* yp = wya + m * 64 + Mb;
    const float* cp = b1a + m * 64 + Mb;
#pragma unroll
    for (int r = 0; r < 8; ++r) {
      b0v[r] = bp[r]; wxv[r] = xp[r]; wyv[r] = yp[r]; b1v[r] = cp[r];
    }
  }

  // ---- layer 0: h0 = relu(W0x * x + (b0 + w0x*lx + w0y*ly)) ----------------
  {
    const v16bf a0 = *(const v16bf*)(w0A + (size_t)m * 4096 +        mt * 512 + lane * 16);
    const v16bf a1 = *(const v16bf*)(w0A + (size_t)m * 4096 + 2048 + mt * 512 + lane * 16);
#pragma unroll
    for (int nt = 0; nt < 4; ++nt) {
      const int col = nt * 16 + lrow;          // pixel within block
      const int pg  = pix0 + col;              // global pixel index
      const float lx = (float)(pg % WW) * (1.0f / (float)WW);
      const float ly = (float)(pg / WW) * (1.0f / (float)WW);
      v8f acc;
#pragma unroll
      for (int r = 0; r < 8; ++r)
        acc[r] = fmaf(wxv[r], lx, fmaf(wyv[r], ly, b0v[r]));

      const v16bf b0 = *(const v16bf*)(&xsT[col * 80 + 16 * half]);
      const v16bf b1 = *(const v16bf*)(&xsT[col * 80 + 16 * half + 32]);
      acc = __builtin_amdgcn_wmma_f32_16x16x32_bf16(false, a0, false, b0, (short)0, acc, false, false);
      acc = __builtin_amdgcn_wmma_f32_16x16x32_bf16(false, a1, false, b1, (short)0, acc, false, false);

      v8bf hv;
#pragma unroll
      for (int r = 0; r < 8; ++r) hv[r] = (bf16_t)fmaxf(acc[r], 0.0f);
      *(v8bf*)(&hsT[col * 80 + Mb]) = hv;      // 16B aligned
    }
  }
  __syncthreads();

  // ---- layer 1: h1 = relu(W1 * h0 + b1) ------------------------------------
  {
    const v16bf a0 = *(const v16bf*)(w1A + (size_t)m * 4096 +        mt * 512 + lane * 16);
    const v16bf a1 = *(const v16bf*)(w1A + (size_t)m * 4096 + 2048 + mt * 512 + lane * 16);
#pragma unroll
    for (int nt = 0; nt < 4; ++nt) {
      const int col = nt * 16 + lrow;
      v8f acc;
#pragma unroll
      for (int r = 0; r < 8; ++r) acc[r] = b1v[r];

      const v16bf b0 = *(const v16bf*)(&hsT[col * 80 + 16 * half]);
      const v16bf b1 = *(const v16bf*)(&hsT[col * 80 + 16 * half + 32]);
      acc = __builtin_amdgcn_wmma_f32_16x16x32_bf16(false, a0, false, b0, (short)0, acc, false, false);
      acc = __builtin_amdgcn_wmma_f32_16x16x32_bf16(false, a1, false, b1, (short)0, acc, false, false);

#pragma unroll
      for (int r = 0; r < 4; ++r) acc[r] = fmaxf(acc[r], 0.0f);
#pragma unroll
      for (int r = 4; r < 8; ++r) acc[r] = fmaxf(acc[r], 0.0f);
      *(float4*)(&hs2[col * 72 + Mb    ]) = make_float4(acc[0], acc[1], acc[2], acc[3]);
      *(float4*)(&hs2[col * 72 + Mb + 4]) = make_float4(acc[4], acc[5], acc[6], acc[7]);
    }
  }
  __syncthreads();

  // ---- layer 2: out = w2 . h1 + (b2 - 2.19), one thread per pixel ----------
  if (t < NPIX) {
    const int pix = t;
    float a = b2a[m];
    const float4* hp = (const float4*)(&hs2[pix * 72]);
    const float4* wp = (const float4*)(w2a + m * 64);  // uniform -> scalar loads
#pragma unroll
    for (int q = 0; q < 16; ++q) {
      const float4 h = hp[q];
      const float4 w = wp[q];
      a = fmaf(w.x, h.x, fmaf(w.y, h.y, fmaf(w.z, h.z, fmaf(w.w, h.w, a))));
    }
    out[(size_t)m * HW + pix0 + pix] = a;
  }
}

// ---------------------------------------------------------------------------
// Host launcher
// ---------------------------------------------------------------------------
extern "C" void kernel_launch(void* const* d_in, const int* in_sizes, int n_in,
                              void* d_out, int out_size, void* d_ws, size_t ws_size,
                              hipStream_t stream) {
  const float* x      = (const float*)d_in[0];   // [4,64,80,200] fp32
  const float* params = (const float*)d_in[1];   // [32,8513] fp32
  float* out          = (float*)d_out;           // [32,16000] fp32

  char* ws = (char*)d_ws;
  bf16_t* w0A = (bf16_t*)(ws);               // 32*4096 bf16 = 262144 B
  bf16_t* w1A = (bf16_t*)(ws + 262144);      // 262144 B
  float*  b0a = (float*)(ws + 524288);       // 8192 B
  float*  wxa = (float*)(ws + 532480);       // 8192 B
  float*  wya = (float*)(ws + 540672);       // 8192 B
  float*  b1a = (float*)(ws + 548864);       // 8192 B
  float*  w2a = (float*)(ws + 557056);       // 8192 B
  float*  b2a = (float*)(ws + 565248);       // 128 B   (total ~565 KB)

  pack_weights_kernel<<<MI, 256, 0, stream>>>(params, w0A, w1A, b0a, wxa, wya, b1a, w2a, b2a);

  dim3 grid(HW / NPIX, MI);   // 250 pixel blocks x 32 instances
  cond_mask_head_kernel<<<grid, 128, 0, stream>>>(x, out, w0A, w1A,
                                                  b0a, wxa, wya, b1a, w2a, b2a);
  (void)in_sizes; (void)n_in; (void)out_size; (void)ws_size;
}
